// SequenceModel_31920196944114
// MI455X (gfx1250) — matrix operand order repaired and checked by reading, not compiled, not run on hardware
//
#include <hip/hip_runtime.h>

// ---------------------------------------------------------------------------
// Tanh-RNN over ragged batch, CDNA5 (gfx1250) WMMA implementation.
//   B=256, T=2048, D_IN=D_H=64, D_OUT=1
// Phase 1: Z[b,t,:] = x[b,t,:] @ W_xh^T   (big parallel GEMM, f16 WMMA, f32 acc)
//          Z stored f16 in d_ws, pre-blocked per (batch-tile, t) in the exact
//          per-lane C-fragment order so phase 2 reads it with 4 b128 loads.
// Phase 2: 16 independent recurrences (16 batch rows each), one wave32 each.
//          h stays in f32 C-fragments; A-operand rebuilt each step via a
//          double-buffered LDS transpose; masked freeze at t >= seq_len;
//          loop truncated at the tile-max seq_len (wave shuffle reduction);
//          hardware v_tanh_f32 when available. W_out projection fused via
//          ds_add_f32 reduction.
// Workspace requirement: 16 tiles * 2048 t * 2KB = 64 MiB in d_ws.
// ---------------------------------------------------------------------------

typedef __attribute__((ext_vector_type(16))) _Float16 v16h;
typedef __attribute__((ext_vector_type(8)))  float    v8f;

#define TT 2048
#define DH 64
#define NTILE 16   // batch tiles of 16 rows

#if __has_builtin(__builtin_amdgcn_tanhf)
#define FAST_TANH(x) __builtin_amdgcn_tanhf(x)   // single v_tanh_f32 (TRANS op)
#else
#define FAST_TANH(x) tanhf(x)
#endif

union FragH {           // 16x32 f16 A/B operand: 16 halves per lane
    v16h     v;
    _Float16 h[16];
    uint4    q[2];
};
union FragF {           // 16x16 f32 C/D: 8 floats per lane
    v8f   v;
    float f[8];
};
union H8 {              // 8 packed halves <-> one b128
    uint4    q;
    _Float16 h[8];
};

__device__ __forceinline__ void cvt4(FragH& f, int dst, float4 w) {
    f.h[dst + 0] = (_Float16)w.x;
    f.h[dst + 1] = (_Float16)w.y;
    f.h[dst + 2] = (_Float16)w.z;
    f.h[dst + 3] = (_Float16)w.w;
}

// Load the 8 B-operand fragments (2 K-halves x 4 N-tiles) of a 64x64 weight
// matrix W[j][i] used as out = in @ W^T  (B element (K=i, N=j) = W[j*64+i]).
// K order chosen here (same as A): lane L holds N = L%16 and the contiguous K range
// [kbase + (L>=16)*16, +16) in halves 0..15  -> 4 float4 loads + convert.
__device__ __forceinline__ void load_b_frags(const float* __restrict__ W,
                                             int ln, int hi, FragH bw[2][4]) {
#pragma unroll
    for (int kk = 0; kk < 2; ++kk)
#pragma unroll
        for (int n = 0; n < 4; ++n) {
            const float* src = W + (ln + 16 * n) * DH + kk * 32 + hi * 16;
#pragma unroll
            for (int q = 0; q < 4; ++q)
                cvt4(bw[kk][n], 4 * q, ((const float4*)src)[q]);
        }
}

// ---------------------------------------------------------------------------
// Phase 1: Z = x @ W_xh^T.  grid = 16 batch-tiles * 64 time-chunks, 1 wave ea.
// A operand (16x32): lane L: M = L%16; halves 0..15 hold K = hi*16+e
// -> one contiguous 16-float run per frag, same K order as B.
// ---------------------------------------------------------------------------
__global__ __launch_bounds__(32)
void rnn_phase1(const float* __restrict__ x, const float* __restrict__ Wxh,
                _Float16* __restrict__ zws) {
    const int lane = threadIdx.x;
    const int ln = lane & 15;
    const int hi = lane >> 4;
    const int k  = blockIdx.x >> 6;          // batch tile 0..15
    const int t0 = (blockIdx.x & 63) * 32;   // time chunk

    FragH bw[2][4];
    load_b_frags(Wxh, ln, hi, bw);

    const int b = k * 16 + ln;               // A row handled by this lane
    const float* xrow = x + (size_t)b * TT * DH;
    uint4* zq = (uint4*)zws;

    for (int tt = 0; tt < 32; ++tt) {
        const int t = t0 + tt;
        const float* xr = xrow + (size_t)t * DH;

        FragH a0, a1;
        cvt4(a0, 0,  ((const float4*)(xr + hi * 16))[0]);
        cvt4(a0, 4,  ((const float4*)(xr + hi * 16))[1]);
        cvt4(a0, 8,  ((const float4*)(xr + hi * 16))[2]);
        cvt4(a0, 12, ((const float4*)(xr + hi * 16))[3]);
        cvt4(a1, 0,  ((const float4*)(xr + 32 + hi * 16))[0]);
        cvt4(a1, 4,  ((const float4*)(xr + 32 + hi * 16))[1]);
        cvt4(a1, 8,  ((const float4*)(xr + 32 + hi * 16))[2]);
        cvt4(a1, 12, ((const float4*)(xr + 32 + hi * 16))[3]);

        FragF c[4];
#pragma unroll
        for (int n = 0; n < 4; ++n) {
            v8f cz = {};
            c[n].v = __builtin_amdgcn_wmma_f32_16x16x32_f16(
                false, a0.v, false, bw[0][n].v, (short)0, cz, false, false);
            c[n].v = __builtin_amdgcn_wmma_f32_16x16x32_f16(
                false, a1.v, false, bw[1][n].v, (short)0, c[n].v, false, false);
        }

        // Pre-blocked f16 store: index ((k*T+t)*4 + n)*32 + lane, 8 halves ea.
        const size_t base = ((size_t)(k * TT + t)) * 4;
#pragma unroll
        for (int n = 0; n < 4; ++n) {
            H8 o;
#pragma unroll
            for (int r = 0; r < 8; ++r) o.h[r] = (_Float16)c[n].f[r];
            zq[(base + n) * 32 + lane] = o.q;
        }
    }
}

// ---------------------------------------------------------------------------
// Phase 2: serial recurrence, one wave32 per 16-row batch tile (16 blocks).
// ---------------------------------------------------------------------------
__global__ __launch_bounds__(32)
void rnn_phase2(const _Float16* __restrict__ zws,
                const int* __restrict__ slen,
                const float* __restrict__ Whh,
                const float* __restrict__ Wout,
                const float* __restrict__ bout,
                float* __restrict__ out) {
    __shared__ _Float16 hbuf[2][16 * DH];   // double-buffered h tile (f16)
    __shared__ float    oacc[16];

    const int lane = threadIdx.x;
    const int ln = lane & 15;
    const int hi = lane >> 4;
    const int k  = blockIdx.x;              // batch tile 0..15

    if (lane < 16) oacc[lane] = 0.0f;

    FragH bw[2][4];
    load_b_frags(Whh, ln, hi, bw);

    int sl[8];
    int mx = 0;
#pragma unroll
    for (int r = 0; r < 8; ++r) {
        sl[r] = slen[k * 16 + r + hi * 8];
        mx = sl[r] > mx ? sl[r] : mx;
    }
    // tile-wide max seq_len: once t >= mx, h is frozen for every row -> stop.
#pragma unroll
    for (int d = 16; d > 0; d >>= 1) {
        int o = __shfl_xor(mx, d, 32);      // ds permute/swizzle, wave32
        mx = o > mx ? o : mx;
    }

    float hst[4][8];
#pragma unroll
    for (int n = 0; n < 4; ++n)
#pragma unroll
        for (int r = 0; r < 8; ++r) hst[n][r] = 0.0f;

    FragH a0, a1;
#pragma unroll
    for (int i = 0; i < 16; ++i) { a0.h[i] = (_Float16)0.0f; a1.h[i] = (_Float16)0.0f; }

    const uint4* zq = (const uint4*)zws;
    uint4 z[4];
#pragma unroll
    for (int n = 0; n < 4; ++n)
        z[n] = zq[((size_t)(k * TT) * 4 + n) * 32 + lane];

    for (int t = 0; t < mx; ++t) {
        // C accumulator initialized with Z tile (prefetched last iteration)
        FragF c[4];
#pragma unroll
        for (int n = 0; n < 4; ++n) {
            H8 zh; zh.q = z[n];
#pragma unroll
            for (int r = 0; r < 8; ++r) c[n].f[r] = (float)zh.h[r];
        }
        // prefetch next step's Z while WMMAs run
        if (t + 1 < mx) {
#pragma unroll
            for (int n = 0; n < 4; ++n)
                z[n] = zq[((size_t)(k * TT + t + 1) * 4 + n) * 32 + lane];
        }

#pragma unroll
        for (int n = 0; n < 4; ++n) {
            c[n].v = __builtin_amdgcn_wmma_f32_16x16x32_f16(
                false, a0.v, false, bw[0][n].v, (short)0, c[n].v, false, false);
            c[n].v = __builtin_amdgcn_wmma_f32_16x16x32_f16(
                false, a1.v, false, bw[1][n].v, (short)0, c[n].v, false, false);
        }

        // tanh + ragged freeze (C/D layout: row M = r + hi*8, col N = ln+16n)
#pragma unroll
        for (int n = 0; n < 4; ++n)
#pragma unroll
            for (int r = 0; r < 8; ++r) {
                float v = FAST_TANH(c[n].f[r]);
                if (t < sl[r]) hst[n][r] = v;
            }

        // D-layout -> A-layout transpose through LDS (double buffered)
        _Float16* hb = hbuf[t & 1];
#pragma unroll
        for (int n = 0; n < 4; ++n)
#pragma unroll
            for (int r = 0; r < 8; ++r)
                hb[(r + hi * 8) * DH + (ln + 16 * n)] = (_Float16)hst[n][r];
        __syncthreads();

        const _Float16* row = hb + ln * DH;   // A row M = ln, K contiguous
        a0.q[0] = *(const uint4*)(row + hi * 16);
        a0.q[1] = *(const uint4*)(row + 8 + hi * 16);
        a1.q[0] = *(const uint4*)(row + 32 + hi * 16);
        a1.q[1] = *(const uint4*)(row + 40 + hi * 16);
    }

    // Fused output projection: out[b] = sum_j h[b][j]*W_out[j] + b_out
    float wo[4];
#pragma unroll
    for (int n = 0; n < 4; ++n) wo[n] = Wout[ln + 16 * n];
#pragma unroll
    for (int r = 0; r < 8; ++r) {
        float acc = 0.0f;
#pragma unroll
        for (int n = 0; n < 4; ++n) acc += hst[n][r] * wo[n];
        atomicAdd(&oacc[r + hi * 8], acc);   // ds_add_f32
    }
    __syncthreads();
    if (lane < 16) out[k * 16 + lane] = oacc[lane] + bout[0];
}

// ---------------------------------------------------------------------------
extern "C" void kernel_launch(void* const* d_in, const int* in_sizes, int n_in,
                              void* d_out, int out_size, void* d_ws, size_t ws_size,
                              hipStream_t stream) {
    const float* x    = (const float*)d_in[0];   // [256,2048,64] f32
    const int*   sl   = (const int*)  d_in[1];   // [256] i32
    const float* Wxh  = (const float*)d_in[2];   // [64,64]
    const float* Whh  = (const float*)d_in[3];   // [64,64]
    const float* Wout = (const float*)d_in[4];   // [1,64]
    const float* bout = (const float*)d_in[5];   // [1]
    _Float16* zws = (_Float16*)d_ws;             // needs 64 MiB scratch
    float* outp = (float*)d_out;                 // [256,1]

    rnn_phase1<<<NTILE * 64, 32, 0, stream>>>(x, Wxh, zws);
    rnn_phase2<<<NTILE, 32, 0, stream>>>(zws, sl, Whh, Wout, bout, outp);
}
